// Decoder_11974368821574
// MI455X (gfx1250) — matrix operand order repaired and checked by reading, not compiled, not run on hardware
//
#include <hip/hip_runtime.h>
#include <hip/hip_bf16.h>
#include <math.h>

// ---------------------------------------------------------------------------
// LSTM decoder + GMM2D log-prob, CDNA5 (gfx1250) wave32 WMMA implementation.
// Sizes: X=256 Z=64 ZX=320 H=64 GC=16 PD=2 B=256 K=16 T=50, N=K*B=4096.
// Grid: 256 workgroups x 128 threads (4 waves). Each WG owns 16 rows of N and
// runs the full 50-step scan privately; wave w owns H columns [16w,16w+16).
//
// Round-2 changes:
//  * A-fragment loads vectorized: 2x ds_load_b128 per fragment (the ISA A
//    layout gives each lane two contiguous 8-half runs, 16B-aligned).
//  * Output-projection B fragments parked in a pre-swizzled LDS buffer
//    (32 contiguous bytes per lane) and reloaded with 2x ds_load_b128 per
//    step, freeing ~32 VGPRs to keep the serial scan loop spill-free.
// ---------------------------------------------------------------------------

typedef __attribute__((ext_vector_type(16))) _Float16 v16h;
typedef __attribute__((ext_vector_type(8)))  _Float16 v8h;
typedef __attribute__((ext_vector_type(8)))  float    v8f;

__device__ __forceinline__ v8f wmma_f16(v16h a, v16h b, v8f c) {
  // D = A(16x32 f16) * B(32x16 f16) + C(16x16 f32)
  return __builtin_amdgcn_wmma_f32_16x16x32_f16(
      /*neg_a=*/false, a, /*neg_b=*/false, b,
      /*c_mod=*/(short)0, c, /*reuse_a=*/false, /*reuse_b=*/false);
}

// A-fragment (16x32 f16) from row-major f16 LDS [16][ld], K-chunk kc.
// ISA 7.12.2: lanes 0-15 hold M=lane; halves 0-7 -> K=kbase..kbase+7,
// halves 8-15 -> K=kbase+16..kbase+23; lanes 16-31 use kbase+8.
// Both runs are contiguous and 16B aligned -> two ds_load_b128 per lane.
__device__ __forceinline__ v16h load_a_frag(const _Float16* lds, int ld,
                                            int kc, int lane) {
  const int m = lane & 15;
  const int kbase = ((lane & 16) ? 8 : 0) + 32 * kc;
  const _Float16* p = lds + m * ld + kbase;
  v8h lo = *(const v8h*)(p);
  v8h hi = *(const v8h*)(p + 16);
  return __builtin_shufflevector(lo, hi, 0, 1, 2, 3, 4, 5, 6, 7, 8, 9, 10, 11,
                                 12, 13, 14, 15);
}

// B-fragment (32x16 f16) of W^T, from row-major f32 global W[ncols][ldw].
// B[k][n] = W[colbase+n][k]. Lanes 0-15: N=lane, K=0..7,16..23; lanes 16-31: K+=8.
// (Compiler coalesces these into global_load_b128 + v_cvt_pk_f16_f32.)
__device__ __forceinline__ v16h load_b_frag(const float* __restrict__ W,
                                            int ldw, int colbase, int kc,
                                            int lane) {
  const int n = lane & 15;
  const int kbase = ((lane & 16) ? 8 : 0) + 32 * kc;
  v16h b;
#pragma unroll
  for (int j = 0; j < 16; ++j)
    b[j] = (_Float16)W[(colbase + n) * ldw + kbase + j + ((j >= 8) ? 8 : 0)];
  return b;
}

__device__ __forceinline__ float sigmoidf_(float x) {
  return 1.0f / (1.0f + __expf(-x));
}

__global__ __launch_bounds__(128) void decoder_lstm_gmm_kernel(
    const float* __restrict__ x, const float* __restrict__ z,
    const float* __restrict__ input_seqs, const float* __restrict__ pred_seqs,
    const float* __restrict__ Wh0, const float* __restrict__ bh0,
    const float* __restrict__ Wc0, const float* __restrict__ bc0,
    const float* __restrict__ Wih, const float* __restrict__ Whh,
    const float* __restrict__ bih, const float* __restrict__ bhh,
    const float* __restrict__ Wpi, const float* __restrict__ bpi,
    const float* __restrict__ Wmu, const float* __restrict__ bmu,
    const float* __restrict__ Wls, const float* __restrict__ bls,
    const float* __restrict__ Wcorr, const float* __restrict__ bcorr,
    float* __restrict__ out) {
  constexpr int T = 50;

  __shared__ __align__(32) _Float16 zx_lds[16 * 320];  // zx rows (f16)
  __shared__ __align__(32) _Float16 h_lds[16 * 64];    // hidden state (f16)
  __shared__ __align__(32) _Float16 pswz_lds[16][32][16];  // proj B-frags
  __shared__ float dec_lds[T][16][2];   // decode inputs per step
  __shared__ float tgt_lds[T][16][2];   // GMM target values per step
  __shared__ float proj_lds[16][96];    // [pi16 | mu32 | ls32 | corr16]
  __shared__ float comp_lds[16][16];    // GMM mixture components scratch

  const int tid = threadIdx.x;
  const int lane = tid & 31;
  const int wave = tid >> 5;
  const int n0 = blockIdx.x * 16;   // first global row (n = k*256 + b)
  const int kk = n0 >> 8;           // z-sample index
  const int b0 = n0 & 255;          // batch base

  // ---------------- Phase 0: stage zx / decode / targets into LDS ----------
  for (int e = tid; e < 16 * 320; e += 128) {
    const int r = e / 320, c = e % 320;
    const float v = (c < 64) ? z[(kk * 256 + b0 + r) * 64 + c]
                             : x[(b0 + r) * 256 + (c - 64)];
    zx_lds[r * 320 + c] = (_Float16)v;
  }
  for (int e = tid; e < T * 16; e += 128) {
    const int t = e / 16, r = e % 16, b = b0 + r;
    float dxv, dyv;
    if (t == 0) {  // tgt_present = input_seqs[b, 7, 1, 1, 2:4]
      dxv = input_seqs[b * 192 + 188];
      dyv = input_seqs[b * 192 + 189];
    } else {       // tgt_future[b, t-1] = pred_seqs[b, t-1, 1, 1, 2:4]
      dxv = pred_seqs[b * 1200 + (t - 1) * 24 + 20];
      dyv = pred_seqs[b * 1200 + (t - 1) * 24 + 21];
    }
    dec_lds[t][r][0] = dxv;
    dec_lds[t][r][1] = dyv;
    tgt_lds[t][r][0] = pred_seqs[b * 1200 + t * 24 + 20];
    tgt_lds[t][r][1] = pred_seqs[b * 1200 + t * 24 + 21];
  }
  __syncthreads();

  // ---------------- Phase 1: h0, c0, gate_zx, resident weight fragments ----
  const int hcb = wave * 16;       // H-column base of this wave
  const int nloc = lane & 15;      // column within 16-wide tile
  const int mbase = (lane & 16) ? 8 : 0;  // C-fragment row base

  // h0 = zx @ Wh0^T + bh0  (written straight to h_lds, no activation)
  {
    v8f acc;
    const float bv = bh0[hcb + nloc];
#pragma unroll
    for (int v = 0; v < 8; ++v) acc[v] = bv;
    for (int kc = 0; kc < 10; ++kc) {
      v16h a = load_a_frag(zx_lds, 320, kc, lane);
      v16h b = load_b_frag(Wh0, 320, hcb, kc, lane);
      acc = wmma_f16(a, b, acc);
    }
#pragma unroll
    for (int v = 0; v < 8; ++v)
      h_lds[(mbase + v) * 64 + hcb + nloc] = (_Float16)acc[v];
  }

  // c0 = zx @ Wc0^T + bc0  (kept in registers as cell state)
  v8f cstate;
  {
    const float bv = bc0[hcb + nloc];
#pragma unroll
    for (int v = 0; v < 8; ++v) cstate[v] = bv;
    for (int kc = 0; kc < 10; ++kc) {
      v16h a = load_a_frag(zx_lds, 320, kc, lane);
      v16h b = load_b_frag(Wc0, 320, hcb, kc, lane);
      cstate = wmma_f16(a, b, cstate);
    }
  }

  // gate_zx[g] = zx @ Wih[:, :320]^T + bih + bhh (time-invariant gate part)
  v8f gz[4];
#pragma unroll
  for (int gt = 0; gt < 4; ++gt) {
    const int gcb = gt * 64 + hcb;
    const float bv = bih[gcb + nloc] + bhh[gcb + nloc];
    v8f acc;
#pragma unroll
    for (int v = 0; v < 8; ++v) acc[v] = bv;
    for (int kc = 0; kc < 10; ++kc) {
      v16h a = load_a_frag(zx_lds, 320, kc, lane);
      v16h b = load_b_frag(Wih, 322, gcb, kc, lane);
      acc = wmma_f16(a, b, acc);
    }
    gz[gt] = acc;
  }

  // Whh^T fragments resident in VGPRs for the whole scan (hot path)
  v16h whhf[4][2];
#pragma unroll
  for (int gt = 0; gt < 4; ++gt)
#pragma unroll
    for (int kc = 0; kc < 2; ++kc)
      whhf[gt][kc] = load_b_frag(Whh, 64, gt * 64 + hcb, kc, lane);

  // rank-2 decode-input weights (Wih columns 320,321), per gate column
  float wdx[4], wdy[4];
#pragma unroll
  for (int gt = 0; gt < 4; ++gt) {
    const int gcol = gt * 64 + hcb + nloc;
    wdx[gt] = Wih[gcol * 322 + 320];
    wdy[gt] = Wih[gcol * 322 + 321];
  }

  // Output-projection tiles. Concat layout P = [pi | mu | ls | corr] (96 cols,
  // 6 tiles). wave0:{0,4} wave1:{1,5} wave2:{2} wave3:{3}. Fragments are
  // parked pre-swizzled in LDS (32 contiguous bytes per lane) and reloaded
  // with 2x ds_load_b128 each step to keep the scan loop spill-free.
  const float* WA;
  const float* bA;
  if (wave == 0)      { WA = Wpi;            bA = bpi;      }
  else if (wave == 1) { WA = Wmu;            bA = bmu;      }
  else if (wave == 2) { WA = Wmu + 16 * 64;  bA = bmu + 16; }
  else                { WA = Wls;            bA = bls;      }
  const float* WB = (wave == 0) ? (Wls + 16 * 64) : Wcorr;
  const float* bB = (wave == 0) ? (bls + 16) : bcorr;
  const bool hasB = (wave < 2);
  const int tiA = wave, tiB = wave + 4;
  const int fA0 = wave * 4 + 0, fA1 = wave * 4 + 1;
  const int fB0 = wave * 4 + 2, fB1 = wave * 4 + 3;

  *(v16h*)&pswz_lds[fA0][lane][0] = load_b_frag(WA, 64, 0, 0, lane);
  *(v16h*)&pswz_lds[fA1][lane][0] = load_b_frag(WA, 64, 0, 1, lane);
  const float pbA = bA[nloc];
  float pbB = 0.0f;
  if (hasB) {
    *(v16h*)&pswz_lds[fB0][lane][0] = load_b_frag(WB, 64, 0, 0, lane);
    *(v16h*)&pswz_lds[fB1][lane][0] = load_b_frag(WB, 64, 0, 1, lane);
    pbB = bB[nloc];
  }

  __syncthreads();  // h_lds(h0) + pswz_lds visible to all waves

  // ---------------- Phase 2: sequential scan over T steps ------------------
  float accum = 0.0f;  // per-row log-prob sum (threads 0..15)

  for (int t = 0; t < T; ++t) {
    // gates = gate_zx + h @ Whh^T + dec_t @ Wih[:,320:322]^T
    v16h ha0 = load_a_frag(h_lds, 64, 0, lane);
    v16h ha1 = load_a_frag(h_lds, 64, 1, lane);
    v8f gat[4];
#pragma unroll
    for (int gt = 0; gt < 4; ++gt) {
      v8f g = gz[gt];
      g = wmma_f16(ha0, whhf[gt][0], g);
      g = wmma_f16(ha1, whhf[gt][1], g);
#pragma unroll
      for (int v = 0; v < 8; ++v) {
        const float ddx = dec_lds[t][mbase + v][0];
        const float ddy = dec_lds[t][mbase + v][1];
        g[v] += ddx * wdx[gt] + ddy * wdy[gt];
      }
      gat[gt] = g;
    }
    // LSTM cell update on C-fragment layout
    v8f hnew;
#pragma unroll
    for (int v = 0; v < 8; ++v) {
      const float iv = sigmoidf_(gat[0][v]);
      const float fv = sigmoidf_(gat[1][v]);
      const float gv = tanhf(gat[2][v]);
      const float ov = sigmoidf_(gat[3][v]);
      const float cv = fv * cstate[v] + iv * gv;
      cstate[v] = cv;
      hnew[v] = ov * tanhf(cv);
    }
    __syncthreads();  // everyone done reading h_lds (and prev GMM finished)
#pragma unroll
    for (int v = 0; v < 8; ++v)
      h_lds[(mbase + v) * 64 + hcb + nloc] = (_Float16)hnew[v];
    __syncthreads();  // new h visible

    // Fused output projections for this step
    v16h pa0 = load_a_frag(h_lds, 64, 0, lane);
    v16h pa1 = load_a_frag(h_lds, 64, 1, lane);
    {
      v8f p;
#pragma unroll
      for (int v = 0; v < 8; ++v) p[v] = pbA;
      p = wmma_f16(pa0, *(const v16h*)&pswz_lds[fA0][lane][0], p);
      p = wmma_f16(pa1, *(const v16h*)&pswz_lds[fA1][lane][0], p);
#pragma unroll
      for (int v = 0; v < 8; ++v)
        proj_lds[mbase + v][tiA * 16 + nloc] = p[v];
    }
    if (hasB) {  // wave-uniform branch: EXEC stays all-ones for WMMA
      v8f p;
#pragma unroll
      for (int v = 0; v < 8; ++v) p[v] = pbB;
      p = wmma_f16(pa0, *(const v16h*)&pswz_lds[fB0][lane][0], p);
      p = wmma_f16(pa1, *(const v16h*)&pswz_lds[fB1][lane][0], p);
      const bool iscorr = (wave == 1);  // tile 5 = corrs -> tanh
#pragma unroll
      for (int v = 0; v < 8; ++v) {
        float pv = p[v];
        if (iscorr) pv = tanhf(pv);
        proj_lds[mbase + v][tiB * 16 + nloc] = pv;
      }
    }
    __syncthreads();  // projections visible

    // GMM2D log-prob for this step: one lane per row
    if (tid < 16) {
      const int r = tid;
      const float vx = tgt_lds[t][r][0];
      const float vy = tgt_lds[t][r][1];
      // log_softmax denominator over log_pis
      float mx = -3.0e38f;
#pragma unroll
      for (int cc = 0; cc < 16; ++cc) mx = fmaxf(mx, proj_lds[r][cc]);
      float se = 0.0f;
#pragma unroll
      for (int cc = 0; cc < 16; ++cc) se += __expf(proj_lds[r][cc] - mx);
      const float lse_pi = mx + __logf(se);

      float cmax = -3.0e38f;
#pragma unroll
      for (int cc = 0; cc < 16; ++cc) {
        const float ls0 = fminf(fmaxf(proj_lds[r][48 + 2 * cc + 0], -10.f), 10.f);
        const float ls1 = fminf(fmaxf(proj_lds[r][48 + 2 * cc + 1], -10.f), 10.f);
        const float mu0 = proj_lds[r][16 + 2 * cc + 0];
        const float mu1 = proj_lds[r][16 + 2 * cc + 1];
        const float rho = proj_lds[r][80 + cc];
        const float z0 = (vx - mu0) * __expf(-ls0);
        const float z1 = (vy - mu1) * __expf(-ls1);
        const float om = 1.0f - rho * rho;
        const float quad = z0 * z0 + z1 * z1 - 2.0f * rho * z0 * z1;
        const float comp = -1.8378770664093453f  // -log(2*pi)
                           - (ls0 + ls1) - 0.5f * __logf(om)
                           - 0.5f * quad / om;
        const float val = (proj_lds[r][cc] - lse_pi) + comp;
        comp_lds[r][cc] = val;
        cmax = fmaxf(cmax, val);
      }
      float s2 = 0.0f;
#pragma unroll
      for (int cc = 0; cc < 16; ++cc) s2 += __expf(comp_lds[r][cc] - cmax);
      const float logp = cmax + __logf(s2);
      accum += fminf(logp, 50.0f);
    }
    // next iteration's barrier (after gates) protects proj/comp reuse
  }

  if (tid < 16) out[n0 + tid] = accum;  // out[K,B] flat == row index n
}

extern "C" void kernel_launch(void* const* d_in, const int* in_sizes, int n_in,
                              void* d_out, int out_size, void* d_ws,
                              size_t ws_size, hipStream_t stream) {
  const float* x          = (const float*)d_in[0];
  const float* z_stacked  = (const float*)d_in[1];
  const float* input_seqs = (const float*)d_in[2];
  const float* pred_seqs  = (const float*)d_in[3];
  const float* Wh0  = (const float*)d_in[4];
  const float* bh0  = (const float*)d_in[5];
  const float* Wc0  = (const float*)d_in[6];
  const float* bc0  = (const float*)d_in[7];
  const float* Wih  = (const float*)d_in[8];
  const float* Whh  = (const float*)d_in[9];
  const float* bih  = (const float*)d_in[10];
  const float* bhh  = (const float*)d_in[11];
  const float* Wpi  = (const float*)d_in[12];
  const float* bpi  = (const float*)d_in[13];
  const float* Wmu  = (const float*)d_in[14];
  const float* bmu  = (const float*)d_in[15];
  const float* Wls  = (const float*)d_in[16];
  const float* bls  = (const float*)d_in[17];
  const float* Wcorr = (const float*)d_in[18];
  const float* bcorr = (const float*)d_in[19];
  float* out = (float*)d_out;

  // N = K*B = 4096 rows, 16 rows per workgroup -> 256 workgroups of 4 waves.
  decoder_lstm_gmm_kernel<<<dim3(256), dim3(128), 0, stream>>>(
      x, z_stacked, input_seqs, pred_seqs, Wh0, bh0, Wc0, bc0, Wih, Whh, bih,
      bhh, Wpi, bpi, Wmu, bmu, Wls, bls, Wcorr, bcorr, out);
}
